// GCNModel_59141699666122
// MI455X (gfx1250) — matrix operand order repaired and checked by reading, not compiled, not run on hardware
//
#include <hip/hip_runtime.h>
#include <hip/hip_bf16.h>

typedef __attribute__((ext_vector_type(16))) _Float16 v16h;
typedef __attribute__((ext_vector_type(8)))  float    v8f;

// ---------------------------------------------------------------------------
// Small utility kernels
// ---------------------------------------------------------------------------
__global__ void fill_kernel(float* __restrict__ p, float v, int n) {
    int i = blockIdx.x * blockDim.x + threadIdx.x;
    if (i < n) p[i] = v;
}

// deg[dst[e]] += 1  (deg pre-filled with 1.0 for the self loop)
__global__ void deg_kernel(const int* __restrict__ dst, float* __restrict__ deg, int E) {
    int e = blockIdx.x * blockDim.x + threadIdx.x;
    if (e < E) atomicAdd(&deg[dst[e]], 1.0f);
}

// dinv = rsqrt(deg)  (deg >= 1 always, self loop);  norm_self = dinv^2 = 1/deg
__global__ void dinv_kernel(const float* __restrict__ deg, float* __restrict__ dinv,
                            float* __restrict__ nself, int n) {
    int i = blockIdx.x * blockDim.x + threadIdx.x;
    if (i < n) {
        float d = rsqrtf(deg[i]);
        dinv[i]  = d;
        nself[i] = d * d;
    }
}

__global__ void edge_norm_kernel(const int* __restrict__ src, const int* __restrict__ dst,
                                 const float* __restrict__ dinv, float* __restrict__ norm, int E) {
    int e = blockIdx.x * blockDim.x + threadIdx.x;
    if (e < E) norm[e] = dinv[src[e]] * dinv[dst[e]];
}

// Pack W[K,OUT] (f32) into per-lane WMMA B fragments (f16).
// frag[(((kc*NT)+nt)*32 + lane)*16 + e] = f16( W[(kc*32 + (lane>>4)*8 + (e&7) + (e>>3)*16)*OUT + nt*16 + (lane&15)] )
// Bijective over K*OUT elements; lets the GEMM load B as one contiguous 32B run per lane.
__global__ void pack_bfrag_kernel(const float* __restrict__ W, _Float16* __restrict__ frag,
                                  int K, int OUT) {
    int idx = blockIdx.x * blockDim.x + threadIdx.x;
    if (idx >= K * OUT) return;
    const int NT = OUT >> 4;
    int e    = idx & 15;
    int lane = (idx >> 4) & 31;
    int rest = idx >> 9;
    int nt   = rest % NT;
    int kc   = rest / NT;
    int k    = kc * 32 + (lane >> 4) * 8 + (e & 7) + (e >> 3) * 16;
    int col  = nt * 16 + (lane & 15);
    frag[idx] = (_Float16)W[(size_t)k * OUT + col];
}

// OUT[i,f] = norm_self[i] * H[i,f]   (self-loop term; also initializes OUT)
__global__ void self_init_kernel(const float* __restrict__ H, const float* __restrict__ ns,
                                 float* __restrict__ OUT, int n, int F, int logF) {
    int gid = blockIdx.x * blockDim.x + threadIdx.x;
    int i = gid >> logF;
    if (i >= n) return;
    int f = gid & (F - 1);
    OUT[(size_t)i * F + f] = ns[i] * H[(size_t)i * F + f];
}

// OUT[dst[e],f] += norm[e] * H[src[e],f]  — coalesced gather + f32 global atomic scatter
__global__ void scatter_kernel(const float* __restrict__ H, const int* __restrict__ src,
                               const int* __restrict__ dst, const float* __restrict__ norm,
                               float* __restrict__ OUT, int E, int F, int logF) {
    int gid = blockIdx.x * blockDim.x + threadIdx.x;
    int e = gid >> logF;
    if (e >= E) return;
    int f = gid & (F - 1);
    float v = norm[e] * H[(size_t)src[e] * F + f];
    atomicAdd(&OUT[(size_t)dst[e] * F + f], v);
}

__global__ void bias_relu_kernel(float* __restrict__ OUT, const float* __restrict__ bias,
                                 int n, int F, int logF) {
    int gid = blockIdx.x * blockDim.x + threadIdx.x;
    int i = gid >> logF;
    if (i >= n) return;
    int f = gid & (F - 1);
    float v = OUT[(size_t)i * F + f] + bias[f];
    OUT[(size_t)i * F + f] = v > 0.0f ? v : 0.0f;
}

// sums[batch[i], f] += H[i, f];  cnts[batch[i]] += 1 (once per node)
__global__ void pool_kernel(const float* __restrict__ H, const int* __restrict__ batch,
                            float* __restrict__ sums, float* __restrict__ cnts,
                            int n, int F, int logF) {
    int gid = blockIdx.x * blockDim.x + threadIdx.x;
    int i = gid >> logF;
    if (i >= n) return;
    int f = gid & (F - 1);
    int g = batch[i];
    atomicAdd(&sums[(size_t)g * F + f], H[(size_t)i * F + f]);
    if (f == 0) atomicAdd(&cnts[g], 1.0f);
}

// out[g,o] = bfc[o] + (1/max(cnt,1)) * sum_f sums[g,f] * Wfc[f,o]
__global__ void fc_kernel(const float* __restrict__ sums, const float* __restrict__ cnts,
                          const float* __restrict__ Wfc, const float* __restrict__ bfc,
                          float* __restrict__ out, int G, int F) {
    int gid = blockIdx.x * blockDim.x + threadIdx.x;
    if (gid >= G * 10) return;
    int g = gid / 10, o = gid % 10;
    float inv = 1.0f / fmaxf(cnts[g], 1.0f);
    float acc = 0.0f;
    for (int f = 0; f < F; ++f)
        acc += sums[(size_t)g * F + f] * Wfc[f * 10 + o];
    out[gid] = bfc[o] + inv * acc;
}

// ---------------------------------------------------------------------------
// WMMA GEMM: H[N,OUT] = X[N,K](f32 -> f16 in regs) @ W(pre-packed B fragments).
// One wave = one 16x16 C tile. No LDS, no barriers:
//  - A fragment (ISA 7.12.2, 16-bit A 16x32): lane L holds row (L&15); lanes
//    0-15 hold K {0..7, 16..23}, lanes 16-31 hold K {8..15, 24..31}. Each run
//    of 8 K-values is contiguous f32 in X -> 2x global_load_b128 + cvt.
//  - B fragment: one contiguous 32B per lane from the pre-packed buffer
//    (a wave reads 1KB fully coalesced).
//  - C (8 VGPRs): lane col = L&15, VGPR v -> row (L>>4)*8 + v.
// ---------------------------------------------------------------------------
template <int OUT, int K>
__global__ __launch_bounds__(256)
void gemm_wmma_kernel(const float* __restrict__ X, const _Float16* __restrict__ Bfrag,
                      float* __restrict__ H, int nrows) {
    constexpr int NT = OUT / 16;
    const int wave = blockIdx.x * (blockDim.x >> 5) + (threadIdx.x >> 5);
    const int mt = wave / NT;
    const int nt = wave % NT;
    if (mt * 16 >= nrows) return;              // uniform per-wave exit

    const int lane = threadIdx.x & 31;
    const int hsel = lane >> 4;
    const int mn   = lane & 15;

    int row = mt * 16 + mn;
    if (row >= nrows) row = nrows - 1;         // clamp (no-op when nrows%16==0)
    const float* __restrict__ arow = X + (size_t)row * K;

    v8f c = {};

#pragma unroll
    for (int kc = 0; kc < K / 32; ++kc) {
        const int ka0 = kc * 32 + hsel * 8;    // first contiguous 8-run
        const int ka1 = ka0 + 16;              // second contiguous 8-run
        float4 a0 = *(const float4*)(arow + ka0);
        float4 a1 = *(const float4*)(arow + ka0 + 4);
        float4 a2 = *(const float4*)(arow + ka1);
        float4 a3 = *(const float4*)(arow + ka1 + 4);

        v16h a;
        a[0]  = (_Float16)a0.x; a[1]  = (_Float16)a0.y;
        a[2]  = (_Float16)a0.z; a[3]  = (_Float16)a0.w;
        a[4]  = (_Float16)a1.x; a[5]  = (_Float16)a1.y;
        a[6]  = (_Float16)a1.z; a[7]  = (_Float16)a1.w;
        a[8]  = (_Float16)a2.x; a[9]  = (_Float16)a2.y;
        a[10] = (_Float16)a2.z; a[11] = (_Float16)a2.w;
        a[12] = (_Float16)a3.x; a[13] = (_Float16)a3.y;
        a[14] = (_Float16)a3.z; a[15] = (_Float16)a3.w;

        v16h b = *(const v16h*)(Bfrag + ((size_t)(kc * NT + nt) * 32 + lane) * 16);

        c = __builtin_amdgcn_wmma_f32_16x16x32_f16(
                /*neg_a=*/false, a, /*neg_b=*/false, b,
                /*c_mod=*/(short)0, c, /*reuse_a=*/false, /*reuse_b=*/false);
    }

    // Store C: per VGPR v, the 16 lanes of each half write 64B contiguous.
    const int rbase = mt * 16 + hsel * 8;
    float* __restrict__ out = H + (size_t)rbase * OUT + nt * 16 + mn;
#pragma unroll
    for (int v = 0; v < 8; ++v) {
        if (rbase + v < nrows) out[(size_t)v * OUT] = c[v];
    }
}

// ---------------------------------------------------------------------------
// Orchestration
// ---------------------------------------------------------------------------
static inline int ceil_div(long long a, int b) { return (int)((a + b - 1) / b); }

extern "C" void kernel_launch(void* const* d_in, const int* in_sizes, int n_in,
                              void* d_out, int out_size, void* d_ws, size_t ws_size,
                              hipStream_t stream) {
    const float* x     = (const float*)d_in[0];
    const int*   ei    = (const int*)  d_in[1];   // [2, E]
    const int*   batch = (const int*)  d_in[2];
    const float* W1 = (const float*)d_in[3];  const float* b1 = (const float*)d_in[4];
    const float* W2 = (const float*)d_in[5];  const float* b2 = (const float*)d_in[6];
    const float* W3 = (const float*)d_in[7];  const float* b3 = (const float*)d_in[8];
    const float* Wfc = (const float*)d_in[9]; const float* bfc = (const float*)d_in[10];

    const int N = in_sizes[2];            // 100000
    const int E = in_sizes[1] / 2;        // 1600000
    const int F = in_sizes[0] / N;        // 128
    const int G = out_size / 10;          // 1024
    const int F1 = in_sizes[4];           // 64  (b1 length)
    const int* srcI = ei;
    const int* dstI = ei + E;

    // --- carve workspace ---------------------------------------------------
    char* w = (char*)d_ws;
    auto carve = [&](size_t bytes) {
        void* p = (void*)w;
        w += (bytes + 255) & ~(size_t)255;
        return p;
    };
    float*    deg    = (float*)   carve((size_t)N * 4);
    float*    dinv   = (float*)   carve((size_t)N * 4);
    float*    nself  = (float*)   carve((size_t)N * 4);
    float*    norm   = (float*)   carve((size_t)E * 4);
    _Float16* Bf1    = (_Float16*)carve((size_t)in_sizes[3] * 2);
    _Float16* Bf2    = (_Float16*)carve((size_t)in_sizes[5] * 2);
    _Float16* Bf3    = (_Float16*)carve((size_t)in_sizes[7] * 2);
    float*    sums   = (float*)   carve((size_t)G * F * 4);
    float*    cnts   = (float*)   carve((size_t)G * 4);
    float*    P      = (float*)   carve((size_t)N * F * 4);   // ping
    float*    Q      = (float*)   carve((size_t)N * F * 4);   // pong
    (void)ws_size; (void)n_in;

    const int B = 256;
    const int log64 = 6, log128 = 7;

    // --- gcn_norm ----------------------------------------------------------
    fill_kernel<<<ceil_div(N, B), B, 0, stream>>>(deg, 1.0f, N);          // self loops
    deg_kernel<<<ceil_div(E, B), B, 0, stream>>>(dstI, deg, E);
    dinv_kernel<<<ceil_div(N, B), B, 0, stream>>>(deg, dinv, nself, N);
    edge_norm_kernel<<<ceil_div(E, B), B, 0, stream>>>(srcI, dstI, dinv, norm, E);

    // --- pre-pack weights into WMMA B fragments (f16) ----------------------
    pack_bfrag_kernel<<<ceil_div(in_sizes[3], B), B, 0, stream>>>(W1, Bf1, F,  F1);
    pack_bfrag_kernel<<<ceil_div(in_sizes[5], B), B, 0, stream>>>(W2, Bf2, F1, F);
    pack_bfrag_kernel<<<ceil_div(in_sizes[7], B), B, 0, stream>>>(W3, Bf3, F,  F);

    const int mtiles = ceil_div(N, 16);
    const long long nf64  = (long long)N * F1;
    const long long nf128 = (long long)N * F;
    const long long ef64  = (long long)E * F1;
    const long long ef128 = (long long)E * F;

    auto gemm_grid = [&](int nt) { return ceil_div((long long)mtiles * nt * 32, B); };

    // --- layer 1: 128 -> 64 ------------------------------------------------
    gemm_wmma_kernel<64, 128><<<gemm_grid(4), B, 0, stream>>>(x, Bf1, P, N);
    self_init_kernel<<<ceil_div(nf64, B), B, 0, stream>>>(P, nself, Q, N, F1, log64);
    scatter_kernel<<<ceil_div(ef64, B), B, 0, stream>>>(P, srcI, dstI, norm, Q, E, F1, log64);
    bias_relu_kernel<<<ceil_div(nf64, B), B, 0, stream>>>(Q, b1, N, F1, log64);

    // --- layer 2: 64 -> 128 ------------------------------------------------
    gemm_wmma_kernel<128, 64><<<gemm_grid(8), B, 0, stream>>>(Q, Bf2, P, N);
    self_init_kernel<<<ceil_div(nf128, B), B, 0, stream>>>(P, nself, Q, N, F, log128);
    scatter_kernel<<<ceil_div(ef128, B), B, 0, stream>>>(P, srcI, dstI, norm, Q, E, F, log128);
    bias_relu_kernel<<<ceil_div(nf128, B), B, 0, stream>>>(Q, b2, N, F, log128);

    // --- layer 3: 128 -> 128 -----------------------------------------------
    gemm_wmma_kernel<128, 128><<<gemm_grid(8), B, 0, stream>>>(Q, Bf3, P, N);
    self_init_kernel<<<ceil_div(nf128, B), B, 0, stream>>>(P, nself, Q, N, F, log128);
    scatter_kernel<<<ceil_div(ef128, B), B, 0, stream>>>(P, srcI, dstI, norm, Q, E, F, log128);
    bias_relu_kernel<<<ceil_div(nf128, B), B, 0, stream>>>(Q, b3, N, F, log128);

    // --- global mean pool + FC --------------------------------------------
    fill_kernel<<<ceil_div(G * F, B), B, 0, stream>>>(sums, 0.0f, G * F);
    fill_kernel<<<ceil_div(G, B), B, 0, stream>>>(cnts, 0.0f, G);
    pool_kernel<<<ceil_div(nf128, B), B, 0, stream>>>(Q, batch, sums, cnts, N, F, log128);
    fc_kernel<<<ceil_div(G * 10, B), B, 0, stream>>>(sums, cnts, Wfc, bfc, (float*)d_out, G, F);
}
